// RNNModel_65936337928923
// MI455X (gfx1250) — matrix operand order repaired and checked by reading, not compiled, not run on hardware
//
#include <hip/hip_runtime.h>
#include <hip/hip_bf16.h>
#include <math.h>

typedef __attribute__((ext_vector_type(16))) _Float16 v16h;
typedef __attribute__((ext_vector_type(8)))  _Float16 v8h;
typedef __attribute__((ext_vector_type(8)))  float    v8f;

#define DEV static __device__ __forceinline__

DEV v8f wmma32(v16h a, v16h b, v8f c) {
  // D = A(16x32,f16) x B(32x16,f16) + C(16x16,f32)
  return __builtin_amdgcn_wmma_f32_16x16x32_f16(false, a, false, b, (short)0, c,
                                                false, false);
}

// gfx1250 hardware tanh (TRANS pipe, co-executes with WMMA). Trailing v_nop
// satisfies the CDNA5 TRANS-op result-use hazard (1 indep op required).
DEV float htanh(float x) {
  float r;
  asm volatile("v_tanh_f32 %0, %1\n\tv_nop" : "=v"(r) : "v"(x));
  return r;
}

// sigmoid via v_exp_f32 (2^x) + v_rcp_f32
DEV float hsig(float x) {
  float e = __builtin_amdgcn_exp2f(x * -1.44269504088896341f);
  return __builtin_amdgcn_rcpf(1.0f + e);
}

DEV v16h cat16(v8h lo, v8h hi) {
  return __builtin_shufflevector(lo, hi, 0, 1, 2, 3, 4, 5, 6, 7, 8, 9, 10, 11,
                                 12, 13, 14, 15);
}

// A fragment: 16x32 tile of row-major A (stride lda elems), origin (m0,k0).
// ISA layout: lane L (L<16): row m0+L, k = {k0..k0+7, k0+16..k0+23};
//             lane L>=16 : row m0+L-16, k = {k0+8..k0+15, k0+24..k0+31}.
DEV v16h load_a(const _Float16* A, int lda, int m0, int k0, int lane) {
  int r = lane & 15, hg = lane >> 4;
  const _Float16* p = A + (size_t)(m0 + r) * lda + k0 + hg * 8;
  return cat16(*(const v8h*)p, *(const v8h*)(p + 16));
}

// B fragment of W^T where W is row-major [N,K] (stride ldw), tile (n0,k0).
// B[k][n] = W[n][k]; lane L holds row n0+(L&15), k = k0+(L>>4)*8 .. +7, k0+16+(L>>4)*8 .. +7.
DEV v16h load_b(const _Float16* W, int ldw, int n0, int k0, int lane) {
  int r = lane & 15, hg = lane >> 4;
  const _Float16* p = W + (size_t)(n0 + r) * ldw + k0 + hg * 8;
  return cat16(*(const v8h*)p, *(const v8h*)(p + 16));
}

// ---------------------------------------------------------------------------
// Generic fused GEMM: C[M,N] = act( sum_p A_p x W_p^T + bias0 + bias1 )
// A_p row-major f16 [M,K_p]; W_p row-major f16 [N,K_p]. Up to 3 pairs (nullable).
// flags bit0: tanh epilogue; bit1: f32 output (else f16).
// 128 threads = 4 waves (2x2); each wave computes a (MI*16)x(NI*16) C tile:
// MI=NI=4 -> 64x64/wave, 16 wmma per 16 b128 loads per k-chunk (1 load/wmma).
// Grid: (N/(2*NI*16), M/(2*MI*16)).
// ---------------------------------------------------------------------------
template <int MI, int NI>
__global__ __launch_bounds__(128) void k_gemm(
    const _Float16* A0, const _Float16* W0, int lda0, int ldw0, int K0,
    const _Float16* A1, const _Float16* W1, int lda1, int ldw1, int K1,
    const _Float16* A2, const _Float16* W2, int lda2, int ldw2, int K2,
    const float* bias0, const float* bias1, void* out, int ldo, int flags) {
  int lane = threadIdx.x & 31, wid = threadIdx.x >> 5;
  int m0 = blockIdx.y * (2 * MI * 16) + (wid & 1) * (MI * 16);
  int n0 = blockIdx.x * (2 * NI * 16) + (wid >> 1) * (NI * 16);
  v8f acc[MI][NI];
#pragma unroll
  for (int i = 0; i < MI; ++i)
#pragma unroll
    for (int j = 0; j < NI; ++j) acc[i][j] = (v8f){};
  const _Float16* As[3] = {A0, A1, A2};
  const _Float16* Ws[3] = {W0, W1, W2};
  int ldas[3] = {lda0, lda1, lda2};
  int ldws[3] = {ldw0, ldw1, ldw2};
  int Ks[3] = {K0, K1, K2};
  for (int p = 0; p < 3; ++p) {
    const _Float16* A = As[p];
    if (!A) continue;
    const _Float16* W = Ws[p];
    int lda = ldas[p], ldw = ldws[p], K = Ks[p];
    for (int k0 = 0; k0 < K; k0 += 32) {
      v16h af[MI], bf[NI];
#pragma unroll
      for (int i = 0; i < MI; ++i) af[i] = load_a(A, lda, m0 + i * 16, k0, lane);
#pragma unroll
      for (int j = 0; j < NI; ++j) bf[j] = load_b(W, ldw, n0 + j * 16, k0, lane);
#pragma unroll
      for (int i = 0; i < MI; ++i)
#pragma unroll
        for (int j = 0; j < NI; ++j) acc[i][j] = wmma32(af[i], bf[j], acc[i][j]);
    }
  }
  int r = lane & 15, hg = lane >> 4;
#pragma unroll
  for (int i = 0; i < MI; ++i)
#pragma unroll
    for (int j = 0; j < NI; ++j) {
      int col = n0 + j * 16 + r;
      float bs = (bias0 ? bias0[col] : 0.f) + (bias1 ? bias1[col] : 0.f);
#pragma unroll
      for (int v = 0; v < 8; ++v) {
        int row = m0 + i * 16 + hg * 8 + v;
        float x = acc[i][j][v] + bs;
        if (flags & 1) x = htanh(x);
        if (flags & 2)
          ((float*)out)[(size_t)row * ldo + col] = x;
        else
          ((_Float16*)out)[(size_t)row * ldo + col] = (_Float16)x;
      }
    }
}

// ---------------------------------------------------------------------------
// Persistent bidirectional GRU layer. grid.x = direction (0 fwd, 1 bwd),
// 1024 threads = 32 waves; wave w owns output columns [48w, 48w+48).
// h[16][512] f16 in LDS, hh[16][1536] f32 in LDS (dynamic shared, 112 KB).
// PRE = x @ Wih^T + bih, precomputed, rows (b*512 + frame).
// Backward: ragged reversal idx = nf-1-s for s<nf else s (involution).
// Output G[16][512][1024]: forward cols 0..511, backward cols 512..1023.
// ---------------------------------------------------------------------------
__global__ __launch_bounds__(1024) void k_gru(
    const _Float16* PREf, const _Float16* PREb, const _Float16* Whhf,
    const _Float16* Whhb, const float* bhhf, const float* bhhb,
    const int* lengths, _Float16* G) {
  extern __shared__ char smem[];
  _Float16* h = (_Float16*)smem;                  // 16*512 f16 = 16 KB
  float* hh = (float*)(smem + 16 * 512 * 2);      // 16*1536 f32 = 96 KB
  const int dir = blockIdx.x;
  const _Float16* PRE = dir ? PREb : PREf;
  const _Float16* Whh = dir ? Whhb : Whhf;
  const float* bhh = dir ? bhhb : bhhf;
  int tid = threadIdx.x;
  for (int i = tid; i < 16 * 512; i += 1024) h[i] = (_Float16)0.f;
  __syncthreads();
  int lane = tid & 31, wid = tid >> 5;
  int n0 = wid * 48;
  for (int s = 0; s < 512; ++s) {
    // hh = h @ Whh^T  (M=16, N=1536, K=512), A fragments straight from LDS
    v8f acc0 = {}, acc1 = {}, acc2 = {};
    for (int k0 = 0; k0 < 512; k0 += 32) {
      v16h a = load_a(h, 512, 0, k0, lane);
      v16h b0 = load_b(Whh, 512, n0, k0, lane);
      v16h b1 = load_b(Whh, 512, n0 + 16, k0, lane);
      v16h b2 = load_b(Whh, 512, n0 + 32, k0, lane);
      acc0 = wmma32(a, b0, acc0);
      acc1 = wmma32(a, b1, acc1);
      acc2 = wmma32(a, b2, acc2);
    }
    int r = lane & 15, hg = lane >> 4;
    v8f accs[3] = {acc0, acc1, acc2};
#pragma unroll
    for (int j = 0; j < 3; ++j) {
      int col = n0 + j * 16 + r;
      float bs = bhh[col];
#pragma unroll
      for (int v = 0; v < 8; ++v) hh[(hg * 8 + v) * 1536 + col] = accs[j][v] + bs;
    }
    __syncthreads();
    // gate phase: each thread owns 8 (m, j) hidden units
    for (int q = 0; q < 8; ++q) {
      int flat = tid + q * 1024;
      int m = flat >> 9, j = flat & 511;
      int nf = lengths[m] >> 3;  // FRAME_RES = 8
      int idx = dir ? ((s < nf) ? nf - 1 - s : s) : s;
      size_t prow = ((size_t)m * 512 + idx) * 1536;
      float pr = (float)PRE[prow + j];
      float pz = (float)PRE[prow + 512 + j];
      float pn = (float)PRE[prow + 1024 + j];
      float hr = hh[m * 1536 + j];
      float hz = hh[m * 1536 + 512 + j];
      float hn = hh[m * 1536 + 1024 + j];
      float rg = hsig(pr + hr);
      float zg = hsig(pz + hz);
      float ng = htanh(pn + rg * hn);
      float hold = (float)h[m * 512 + j];
      float hnew = (1.f - zg) * ng + zg * hold;
      h[m * 512 + j] = (_Float16)hnew;
      G[((size_t)m * 512 + idx) * 1024 + (size_t)dir * 512 + j] = (_Float16)hnew;
    }
    __syncthreads();
  }
}

// --------------------------- small helper kernels ---------------------------
__global__ void k_cvt(const float* s, _Float16* d, int n) {
  int i = blockIdx.x * blockDim.x + threadIdx.x;
  if (i < n) d[i] = (_Float16)s[i];
}

__global__ void k_fcw_pad(const float* s, _Float16* d) {  // [61,1024] -> [64,1024]
  int i = blockIdx.x * blockDim.x + threadIdx.x;
  if (i < 64 * 1024) d[i] = (i >> 10) < 61 ? (_Float16)s[i] : (_Float16)0.f;
}

__global__ void k_fcb_pad(const float* s, float* d) {
  int i = threadIdx.x;
  if (i < 64) d[i] = i < 61 ? s[i] : 0.f;
}

// batch[16,4096,128] f32 -> X16[t=8][m=8192][128] f16 (t-major step slices)
__global__ void k_frames(const float* batch, _Float16* X, int n) {
  int i = blockIdx.x * blockDim.x + threadIdx.x;
  if (i >= n) return;
  int c = i & 127;
  int m = (i >> 7) & 8191;
  int t = i >> 20;
  int b = m >> 9, f = m & 511;
  X[i] = (_Float16)batch[((size_t)b * 4096 + f * 8 + t) * 128 + c];
}

__global__ void k_offs(const int* lengths, int* offs) {
  if (threadIdx.x == 0 && blockIdx.x == 0) {
    int a = 0;
    for (int b = 0; b < 16; ++b) { offs[b] = a; a += lengths[b] >> 3; }
  }
}

__global__ void k_gather(const float* logits, const int* lengths,
                         const int* offs, float* out, int total) {
  int i = blockIdx.x * blockDim.x + threadIdx.x;
  if (i >= total) return;
  int o = i % 61;
  int bf = i / 61;
  int f = bf & 511, b = bf >> 9;
  if (f < (lengths[b] >> 3))
    out[((size_t)(offs[b] + f)) * 61 + o] = logits[(size_t)bf * 64 + o];
}

// ---------------------------------------------------------------------------
extern "C" void kernel_launch(void* const* d_in, const int* in_sizes, int n_in,
                              void* d_out, int out_size, void* d_ws,
                              size_t ws_size, hipStream_t stream) {
  if (n_in < 36) return;
  const float* batch = (const float*)d_in[0];
  const float* r1l1[8]; const float* r1l2[8];
  const float* r2l1[8]; const float* r2l2[8];
  for (int i = 0; i < 8; ++i) {
    r1l1[i] = (const float*)d_in[1 + i];
    r1l2[i] = (const float*)d_in[9 + i];
    r2l1[i] = (const float*)d_in[17 + i];
    r2l2[i] = (const float*)d_in[25 + i];
  }
  const float* fcW = (const float*)d_in[33];
  const float* fcb = (const float*)d_in[34];
  const int* lengths = (const int*)d_in[35];

  char* ws = (char*)d_ws;
  size_t off = 0;
  auto alloc = [&](size_t bytes) -> void* {
    void* p = ws + off;
    off = (off + bytes + 255) & ~(size_t)255;
    return p;
  };
  // f16 weights
  _Float16* w1ih[2] = {(_Float16*)alloc(512 * 128 * 2), nullptr};
  _Float16* w1hh[2] = {(_Float16*)alloc(512 * 512 * 2), nullptr};
  w1ih[1] = (_Float16*)alloc(512 * 128 * 2);
  w1hh[1] = (_Float16*)alloc(512 * 512 * 2);
  _Float16* w2ih[2] = {(_Float16*)alloc(512 * 1024 * 2), nullptr};
  _Float16* w2hh[2] = {(_Float16*)alloc(512 * 512 * 2), nullptr};
  w2ih[1] = (_Float16*)alloc(512 * 1024 * 2);
  w2hh[1] = (_Float16*)alloc(512 * 512 * 2);
  _Float16* g1ih[2] = {(_Float16*)alloc(1536 * 1024 * 2), nullptr};
  _Float16* g1hh[2] = {(_Float16*)alloc(1536 * 512 * 2), nullptr};
  g1ih[1] = (_Float16*)alloc(1536 * 1024 * 2);
  g1hh[1] = (_Float16*)alloc(1536 * 512 * 2);
  _Float16* g2ih[2] = {(_Float16*)alloc(1536 * 1024 * 2), nullptr};
  _Float16* g2hh[2] = {(_Float16*)alloc(1536 * 512 * 2), nullptr};
  g2ih[1] = (_Float16*)alloc(1536 * 1024 * 2);
  g2hh[1] = (_Float16*)alloc(1536 * 512 * 2);
  _Float16* fcw16 = (_Float16*)alloc(64 * 1024 * 2);
  float* fcb64 = (float*)alloc(64 * 4);
  // activations
  const size_t XS = (size_t)8192 * 128;   // per-step X slice
  const size_t SL = (size_t)8192 * 512;   // per-step h slice
  _Float16* X16 = (_Float16*)alloc(8 * XS * 2);
  _Float16* S1f = (_Float16*)alloc(8 * SL * 2);
  _Float16* S1b = (_Float16*)alloc(8 * SL * 2);
  _Float16* hpp = (_Float16*)alloc(2 * SL * 2);
  _Float16* seq2 = (_Float16*)alloc((size_t)8192 * 1024 * 2);
  _Float16* PREf = (_Float16*)alloc((size_t)8192 * 1536 * 2);
  _Float16* PREb = (_Float16*)alloc((size_t)8192 * 1536 * 2);
  _Float16* G1 = (_Float16*)alloc((size_t)8192 * 1024 * 2);
  _Float16* G2 = (_Float16*)alloc((size_t)8192 * 1024 * 2);
  float* logits = (float*)alloc((size_t)8192 * 64 * 4);
  int* offs = (int*)alloc(32 * 4);

  auto cvt = [&](const float* s, _Float16* d, int n) {
    k_cvt<<<(n + 255) / 256, 256, 0, stream>>>(s, d, n);
  };
  cvt(r1l1[0], w1ih[0], 512 * 128);  cvt(r1l1[1], w1hh[0], 512 * 512);
  cvt(r1l1[4], w1ih[1], 512 * 128);  cvt(r1l1[5], w1hh[1], 512 * 512);
  cvt(r1l2[0], w2ih[0], 512 * 1024); cvt(r1l2[1], w2hh[0], 512 * 512);
  cvt(r1l2[4], w2ih[1], 512 * 1024); cvt(r1l2[5], w2hh[1], 512 * 512);
  cvt(r2l1[0], g1ih[0], 1536 * 1024); cvt(r2l1[1], g1hh[0], 1536 * 512);
  cvt(r2l1[4], g1ih[1], 1536 * 1024); cvt(r2l1[5], g1hh[1], 1536 * 512);
  cvt(r2l2[0], g2ih[0], 1536 * 1024); cvt(r2l2[1], g2hh[0], 1536 * 512);
  cvt(r2l2[4], g2ih[1], 1536 * 1024); cvt(r2l2[5], g2hh[1], 1536 * 512);
  k_fcw_pad<<<(64 * 1024 + 255) / 256, 256, 0, stream>>>(fcW, fcw16);
  k_fcb_pad<<<1, 64, 0, stream>>>(fcb, fcb64);
  k_frames<<<(8 * 8192 * 128 + 255) / 256, 256, 0, stream>>>(batch, X16,
                                                             8 * 8192 * 128);

  // N % 128 == 0 -> big variant (64x64/wave); else small (32x32/wave).
  auto gemm = [&](const _Float16* A0, const _Float16* W0, int la0, int lw0, int K0,
                  const _Float16* A1, const _Float16* W1, int la1, int lw1, int K1,
                  const _Float16* A2, const _Float16* W2, int la2, int lw2, int K2,
                  const float* b0, const float* b1, void* out, int ldo, int flags,
                  int M, int N) {
    if (N % 128 == 0) {
      dim3 g(N / 128, M / 128);
      k_gemm<4, 4><<<g, 128, 0, stream>>>(A0, W0, la0, lw0, K0, A1, W1, la1,
                                          lw1, K1, A2, W2, la2, lw2, K2, b0,
                                          b1, out, ldo, flags);
    } else {
      dim3 g(N / 64, M / 64);
      k_gemm<2, 2><<<g, 128, 0, stream>>>(A0, W0, la0, lw0, K0, A1, W1, la1,
                                          lw1, K1, A2, W2, la2, lw2, K2, b0,
                                          b1, out, ldo, flags);
    }
  };

  // rnn1 layer 1 forward: h_t = tanh(x_t Wih^T + h_{t-1} Whh^T + bih + bhh)
  for (int t = 0; t < 8; ++t)
    gemm(X16 + t * XS, w1ih[0], 128, 128, 128,
         t ? S1f + (t - 1) * SL : nullptr, w1hh[0], 512, 512, 512,
         nullptr, nullptr, 0, 0, 0,
         r1l1[2], r1l1[3], S1f + t * SL, 512, 1, 8192, 512);
  // rnn1 layer 1 backward (stored in original time order)
  for (int t = 7; t >= 0; --t)
    gemm(X16 + t * XS, w1ih[1], 128, 128, 128,
         (t < 7) ? S1b + (t + 1) * SL : nullptr, w1hh[1], 512, 512, 512,
         nullptr, nullptr, 0, 0, 0,
         r1l1[6], r1l1[7], S1b + t * SL, 512, 1, 8192, 512);
  // rnn1 layer 2 forward; only t=7 output is kept (-> seq2 cols 0..511)
  for (int t = 0; t < 8; ++t) {
    const _Float16* prev = t ? hpp + ((t - 1) & 1) * SL : nullptr;
    _Float16* outp = (t < 7) ? hpp + (t & 1) * SL : seq2;
    int ldo = (t < 7) ? 512 : 1024;
    gemm(S1f + t * SL, w2ih[0], 512, 1024, 512,
         S1b + t * SL, w2ih[0] + 512, 512, 1024, 512,
         prev, w2hh[0], 512, 512, 512,
         r1l2[2], r1l2[3], outp, ldo, 1, 8192, 512);
  }
  // rnn1 layer 2 backward: output at t=7 is backward scan step 0 (h0=0)
  gemm(S1f + 7 * SL, w2ih[1], 512, 1024, 512,
       S1b + 7 * SL, w2ih[1] + 512, 512, 1024, 512,
       nullptr, nullptr, 0, 0, 0,
       r1l2[6], r1l2[7], seq2 + 512, 1024, 1, 8192, 512);

  size_t shmem = (size_t)16 * 512 * 2 + (size_t)16 * 1536 * 4;
  // GRU layer 1
  gemm(seq2, g1ih[0], 1024, 1024, 1024, nullptr, nullptr, 0, 0, 0,
       nullptr, nullptr, 0, 0, 0, r2l1[2], nullptr, PREf, 1536, 0, 8192, 1536);
  gemm(seq2, g1ih[1], 1024, 1024, 1024, nullptr, nullptr, 0, 0, 0,
       nullptr, nullptr, 0, 0, 0, r2l1[6], nullptr, PREb, 1536, 0, 8192, 1536);
  k_gru<<<2, 1024, shmem, stream>>>(PREf, PREb, g1hh[0], g1hh[1], r2l1[3],
                                    r2l1[7], lengths, G1);
  // GRU layer 2
  gemm(G1, g2ih[0], 1024, 1024, 1024, nullptr, nullptr, 0, 0, 0,
       nullptr, nullptr, 0, 0, 0, r2l2[2], nullptr, PREf, 1536, 0, 8192, 1536);
  gemm(G1, g2ih[1], 1024, 1024, 1024, nullptr, nullptr, 0, 0, 0,
       nullptr, nullptr, 0, 0, 0, r2l2[6], nullptr, PREb, 1536, 0, 8192, 1536);
  k_gru<<<2, 1024, shmem, stream>>>(PREf, PREb, g2hh[0], g2hh[1], r2l2[3],
                                    r2l2[7], lengths, G2);
  // FC (N padded 61 -> 64), f32 logits (small-tile variant)
  gemm(G2, fcw16, 1024, 1024, 1024, nullptr, nullptr, 0, 0, 0,
       nullptr, nullptr, 0, 0, 0, fcb64, nullptr, logits, 64, 2, 8192, 64);
  // ragged gather
  k_offs<<<1, 1, 0, stream>>>(lengths, offs);
  int total = 16 * 512 * 61;
  k_gather<<<(total + 255) / 256, 256, 0, stream>>>(logits, lengths, offs,
                                                    (float*)d_out, total);
}
